// DepthConv2d_75763223101618
// MI455X (gfx1250) — compile-verified, matrix operand order checked by reading, not run on hardware
//
#include <hip/hip_runtime.h>

typedef __attribute__((ext_vector_type(16))) __bf16  v16bf;
typedef __attribute__((ext_vector_type(8)))  float   v8f;
typedef __attribute__((ext_vector_type(4)))  float   v4f;

#define CIN_   64
#define COUT_  64
#define H_     128
#define BATCH_ 8
#define ALPHA_ 8.3f

#define XPAD   130          // x = -1 .. 128 -> index 0..129
#define CSTR   72           // strip row stride in bf16 (144 B, bank-friendly)
#define WSB_BYTES (18 * 4 * 32 * 8 * 4)   // 73728 B packed B fragments

// f32 -> bf16 round-to-nearest-even
__device__ __forceinline__ unsigned int f32_to_bf16_bits(float f) {
    unsigned int u = __float_as_uint(f);
    u += 0x7fffu + ((u >> 16) & 1u);
    return u >> 16;
}

// ---------------------------------------------------------------------------
// Kernel 1: pack weight[COUT][CIN][3][3] (f32) into bf16 B fragments.
// ws dwords: [chunk = kk*2+sub (0..17)][nt 0..3][lane 0..31][v 0..7]
// B fragment (32x16 16-bit) per-lane layout (mirror of A layout, transposed):
//   lane<16  : col N=lane,    v0..3 = K(0..7) pairs,  v4..7 = K(16..23)
//   lane>=16 : col N=lane-16, v0..3 = K(8..15) pairs, v4..7 = K(24..31)
// K dimension = input channel c within a 32-chunk (sub selects chunk).
__global__ void pack_weight_bf16(const float* __restrict__ w,
                                 unsigned int* __restrict__ ws) {
    int t = blockIdx.x * blockDim.x + threadIdx.x;      // 0 .. 18431
    if (t >= 18 * 4 * 32 * 8) return;
    int v     = t & 7;
    int lane  = (t >> 3) & 31;
    int nt    = (t >> 8) & 3;
    int chunk = t >> 10;                                // kk*2 + sub
    int kk    = chunk >> 1;
    int sub   = chunk & 1;
    int o     = nt * 16 + (lane & 15);
    int koff  = (v < 4) ? (2 * v) : (16 + 2 * (v - 4));
    if (lane >= 16) koff += 8;
    int c0 = sub * 32 + koff;                           // even, < 64
    unsigned int pack = 0;
    #pragma unroll
    for (int e = 0; e < 2; ++e) {
        int c = c0 + e;
        float f = w[(o * CIN_ + c) * 9 + kk];
        pack |= f32_to_bf16_bits(f) << (16 * e);
    }
    ws[t] = pack;
}

// ---------------------------------------------------------------------------
// Kernel 2: one workgroup per (b, oh).
// out[l,o] = bias[o] + sum_kk sim[l,kk] * sum_c strip[ki][l+kj][c] * W[o,c,kk]
__global__ __launch_bounds__(256)
void depthconv_wmma(const float* __restrict__ img,
                    const float* __restrict__ depth,
                    const unsigned int* __restrict__ wsB,
                    const float* __restrict__ bias,
                    float* __restrict__ out) {
    __shared__ float        sim[9][128];                     //  4.6 KB
    __shared__ unsigned int strip[3 * XPAD * (CSTR / 2)];    // 56.2 KB

    const int oh  = blockIdx.x;
    const int b   = blockIdx.y;
    const int tid = threadIdx.x;

    // warm L2/L0 with the packed-B table while we stage LDS (global_prefetch)
    for (int off = tid * 128; off < WSB_BYTES; off += 256 * 128)
        __builtin_prefetch((const char*)wsB + off, 0, 3);

    // ---- phase 0a: modulation table sim[kk][ow]
    const float* dpl = depth + (size_t)b * H_ * H_;
    for (int t = tid; t < 9 * 128; t += 256) {
        int kk = t >> 7;
        int l  = t & 127;
        int ki = kk / 3, kj = kk - 3 * ki;
        int y = oh + ki - 1, x = l + kj - 1;
        float dv = ((unsigned)y < H_ && (unsigned)x < H_) ? dpl[y * H_ + x] : 0.f;
        float ce = (oh >= 1 && l >= 1) ? dpl[(oh - 1) * H_ + (l - 1)] : 0.f;
        sim[kk][l] = __expf(-ALPHA_ * fabsf(dv - ce));
    }

    // ---- phase 0b: raw bf16 image strip, transposed: strip[ki][x'][c]
    // x' = 0..129 covers img x = -1..128 (zero padded); c contiguous.
    // iteration order: x fastest -> coalesced global loads along x.
    const float* imgb = img + (size_t)b * CIN_ * H_ * H_;
    for (int t = tid; t < 3 * 32 * XPAD; t += 256) {
        int x  = t % XPAD;                  // 0..129
        int q  = t / XPAD;
        int cp = q & 31;                    // c pair: c = 2*cp, 2*cp+1
        int ki = q >> 5;                    // 0..2
        int y    = oh + ki - 1;
        int imgx = x - 1;
        unsigned int pack = 0;
        if ((unsigned)y < H_ && (unsigned)imgx < H_) {
            const float* p = imgb + ((2 * cp) * H_ + y) * H_ + imgx;
            pack  = f32_to_bf16_bits(p[0]);
            pack |= f32_to_bf16_bits(p[H_ * H_]) << 16;
        }
        strip[(ki * XPAD + x) * (CSTR / 2) + cp] = pack;
    }
    __syncthreads();

    const int wave  = tid >> 5;
    const int lane  = tid & 31;
    const int half  = lane >> 4;                 // K-half select for A/B frags
    const int arow  = wave * 16 + (lane & 15);   // output row l for A frag

    v8f acc[4] = {};                        // final 4 N-tiles of 16x16 f32
    const v8f zero = {};

    for (int kk = 0; kk < 9; ++kk) {
        int ki = kk / 3, kj = kk - 3 * ki;

        v8f part[4];
        #pragma unroll
        for (int sub = 0; sub < 2; ++sub) { // channel chunks of 32
            // A fragment: row l=arow, K = c in [sub*32, sub*32+32)
            union { uint4 q[2]; v16bf v; } au;
            const __bf16* ab = (const __bf16*)strip
                             + (ki * XPAD + arow + kj) * CSTR
                             + sub * 32 + half * 8;
            au.q[0] = *(const uint4*)(ab);        // c 0..7   (or 8..15)
            au.q[1] = *(const uint4*)(ab + 16);   // c 16..23 (or 24..31)

            union { uint4 q[2]; v16bf v; } bu;
            const int chunk = kk * 2 + sub;
            #pragma unroll
            for (int nt = 0; nt < 4; ++nt) {
                const uint4* bp =
                    (const uint4*)(wsB + (((chunk * 4 + nt) * 32) + lane) * 8);
                bu.q[0] = bp[0];
                bu.q[1] = bp[1];
                part[nt] = __builtin_amdgcn_wmma_f32_16x16x32_bf16(
                    false, au.v, false, bu.v, (short)0,
                    (sub == 0) ? zero : part[nt], false, false);
            }
        }

        // scale partial GEMM by sim[l, kk] (per output row) and accumulate.
        // C layout: VGPR i holds row M = i (lanes<16) / 8+i (lanes>=16).
        const v4f* sp = (const v4f*)&sim[kk][wave * 16 + half * 8];
        v4f s0 = sp[0], s1 = sp[1];
        float sv[8] = { s0.x, s0.y, s0.z, s0.w, s1.x, s1.y, s1.z, s1.w };
        #pragma unroll
        for (int nt = 0; nt < 4; ++nt)
            #pragma unroll
            for (int i = 0; i < 8; ++i)
                acc[nt][i] = fmaf(sv[i], part[nt][i], acc[nt][i]);
    }

    // ---- epilogue: lane = N col (o), VGPR i = M row (ow).
    // Output is write-once: non-temporal stores keep image/B resident in L2.
    const int n = lane & 15;
    #pragma unroll
    for (int nt = 0; nt < 4; ++nt) {
        int o = nt * 16 + n;
        float bv = bias[o];
        float* p = out + ((((size_t)b * COUT_ + o) * H_) + oh) * H_
                       + wave * 16 + half * 8;
        v4f lo, hi;
        lo.x = acc[nt][0] + bv; lo.y = acc[nt][1] + bv;
        lo.z = acc[nt][2] + bv; lo.w = acc[nt][3] + bv;
        hi.x = acc[nt][4] + bv; hi.y = acc[nt][5] + bv;
        hi.z = acc[nt][6] + bv; hi.w = acc[nt][7] + bv;
        __builtin_nontemporal_store(lo, (v4f*)(p));
        __builtin_nontemporal_store(hi, (v4f*)(p + 4));
    }
}

// ---------------------------------------------------------------------------
extern "C" void kernel_launch(void* const* d_in, const int* in_sizes, int n_in,
                              void* d_out, int out_size, void* d_ws, size_t ws_size,
                              hipStream_t stream) {
    (void)in_sizes; (void)n_in; (void)out_size; (void)ws_size;
    const float* image  = (const float*)d_in[0];
    const float* depth  = (const float*)d_in[1];
    const float* weight = (const float*)d_in[2];
    const float* bias   = (const float*)d_in[3];
    float*        outp  = (float*)d_out;
    unsigned int* wsB   = (unsigned int*)d_ws;   // 18432 dwords = 72 KB

    pack_weight_bf16<<<72, 256, 0, stream>>>(weight, wsB);

    dim3 grid(H_, BATCH_);
    depthconv_wmma<<<grid, 256, 0, stream>>>(image, depth, wsB, bias, outp);
}
